// FactorisedPCConv_28501402976579
// MI455X (gfx1250) — matrix operand order repaired and checked by reading, not compiled, not run on hardware
//
#include <hip/hip_runtime.h>

// ---------------------------------------------------------------------------
// FactorisedPCConv fused kernel for MI455X (gfx1250, wave32, WMMA bf16)
//
//   out = segment_sum( (x[src] * MLP(rel_pos)) , dst ) @ Wout + bout
// Fused as:
//   out init= bout;  per edge-tile: atomicAdd( (x[src]*kern) @ Wout )
// so the 12.8MB output and 12.8MB x table both stay L2-resident and only the
// irreducible 64 f32 atomics/edge touch the cache atomic units.
// ---------------------------------------------------------------------------

#define C_IN   64
#define C_OUT  64
#define HID    32
#define WPB    8              // waves per block (wave32)
#define BLOCK  (WPB * 32)

typedef __attribute__((ext_vector_type(16))) __bf16 v16bf;
typedef __attribute__((ext_vector_type(8)))  float  v8f;

__global__ void pcconv_init_out(float* __restrict__ out,
                                const float* __restrict__ bout, int total) {
  int i = blockIdx.x * blockDim.x + threadIdx.x;
  if (i < total) out[i] = bout[i & (C_OUT - 1)];
}

__global__ __launch_bounds__(BLOCK)
void pcconv_edge_fused(const float* __restrict__ x,
                       const float* __restrict__ rel_pos,
                       const int*   __restrict__ src_idx,
                       const int*   __restrict__ dst_idx,
                       const float* __restrict__ W1,
                       const float* __restrict__ b1,
                       const float* __restrict__ W2,
                       const float* __restrict__ b2,
                       const float* __restrict__ Wout,
                       float* __restrict__ out,
                       int n_edges) {
  // ---- block-shared parameter staging ----
  __shared__ float  sW1[3 * HID];
  __shared__ float  sb1[HID];
  __shared__ float  sb2[C_IN];
  __shared__ __bf16 sW2[HID * C_IN];     // [hid][c_in]
  __shared__ __bf16 sWout[C_IN * C_OUT]; // [c_in][c_out]
  // ---- per-wave scratch (no cross-wave sync needed) ----
  __shared__ __bf16 sH[WPB][16 * HID];    // 16 edges x 32 hidden
  __shared__ __bf16 sMsg[WPB][16 * C_IN]; // 16 edges x 64 channels
  __shared__ int    sSrc[WPB][16];
  __shared__ int    sDst[WPB][16];

  const int tid = threadIdx.x;
  for (int i = tid; i < 3 * HID; i += BLOCK)       sW1[i]   = W1[i];
  for (int i = tid; i < HID; i += BLOCK)           sb1[i]   = b1[i];
  for (int i = tid; i < C_IN; i += BLOCK)          sb2[i]   = b2[i];
  for (int i = tid; i < HID * C_IN; i += BLOCK)    sW2[i]   = (__bf16)W2[i];
  for (int i = tid; i < C_IN * C_OUT; i += BLOCK)  sWout[i] = (__bf16)Wout[i];
  __syncthreads();

  const int wave = tid >> 5;
  const int lane = tid & 31;
  const int n    = lane & 15;   // tile column (B/C/D layout: N = lane&15)
  const int hi   = lane >> 4;   // lane half (selects K-group / row-group)

  // ---- preload B fragments into registers (16-bit B 32x16, contiguous k order) ----
  // B element j of v16bf <-> K = 16*hi + j (VGPR j/2, half j&1), N = n.
  v16bf w2B[4];                     // kern GEMM: B = W2 (32 x 64), 4 N-tiles
#pragma unroll
  for (int t = 0; t < 4; ++t)
#pragma unroll
    for (int j = 0; j < 16; ++j)
      w2B[t][j] = sW2[(16 * hi + j) * C_IN + 16 * t + n];

  v16bf woB[2][4];                  // out GEMM: B = Wout (64 x 64), 2 K-halves
#pragma unroll
  for (int k = 0; k < 2; ++k)
#pragma unroll
    for (int t = 0; t < 4; ++t)
#pragma unroll
      for (int j = 0; j < 16; ++j)
        woB[k][t][j] = sWout[(32 * k + 16 * hi + j) * C_OUT + 16 * t + n];

  __bf16* myH   = sH[wave];
  __bf16* myM   = sMsg[wave];
  int*    mySrc = sSrc[wave];
  int*    myDst = sDst[wave];

  const int ntiles     = (n_edges + 15) >> 4;
  const int totalWaves = gridDim.x * WPB;
  for (int tile = blockIdx.x * WPB + wave; tile < ntiles; tile += totalWaves) {
    const int ebase = tile << 4;

    // speculative prefetch of next tile's coordinates (global_prefetch_b8)
    {
      int nt = tile + totalWaves;
      if (nt < ntiles && lane < 3)
        __builtin_prefetch(&rel_pos[(size_t)nt * 48 + lane * 16], 0, 0);
    }

    // ---- phase 1: stage indices, tiny 3->32 MLP layer (scalar VALU) ----
    {
      const int  m    = lane & 15;
      const int  e    = ebase + m;
      const bool v_ok = (e < n_edges);
      if (hi == 0) {
        mySrc[m] = v_ok ? src_idx[e] : 0;
        myDst[m] = v_ok ? dst_idx[e] : -1;
      }
      float px = 0.f, py = 0.f, pz = 0.f;
      if (v_ok) {
        px = rel_pos[e * 3 + 0];
        py = rel_pos[e * 3 + 1];
        pz = rel_pos[e * 3 + 2];
      }
      // each lane-half computes 16 hidden units of its edge
#pragma unroll
      for (int j = 0; j < 16; ++j) {
        const int J = 16 * hi + j;
        float h = fmaf(px, sW1[0 * HID + J],
                  fmaf(py, sW1[1 * HID + J],
                  fmaf(pz, sW1[2 * HID + J], sb1[J])));
        h = h > 0.f ? h : 0.f;
        myH[m * HID + J] = (__bf16)h;
      }
    }
    __builtin_amdgcn_wave_barrier();

    // ---- load A fragment of h (16x32 bf16, contiguous k order matching B) ----
    // A element j <-> row M = lane&15, K = 16*hi + j
    v16bf hA;
#pragma unroll
    for (int j = 0; j < 16; ++j) {
      const int K = 16 * hi + j;
      hA[j] = myH[n * HID + K];
    }

    // ---- phase 2: kern = h @ W2 + b2 (4 WMMAs), msgs = x[src]*kern ----
    // D layout: element v <-> row m = 8*hi+v, col c = 16*t+n
#pragma unroll
    for (int t = 0; t < 4; ++t) {
      const float bias = sb2[16 * t + n];
      v8f kacc = {bias, bias, bias, bias, bias, bias, bias, bias};
      kacc = __builtin_amdgcn_wmma_f32_16x16x32_bf16(
          false, hA, false, w2B[t], (short)0, kacc, false, false);
#pragma unroll
      for (int v = 0; v < 8; ++v) {
        const int m  = 8 * hi + v;
        const int s  = mySrc[m];
        const float xv = x[(size_t)s * C_IN + 16 * t + n];  // L2-resident gather
        myM[m * C_IN + 16 * t + n] = (__bf16)(xv * kacc[v]);
      }
    }
    __builtin_amdgcn_wave_barrier();

    // ---- load msgs A fragments (16x64 -> two 16x32 K-halves) ----
    v16bf mA[2];
#pragma unroll
    for (int k = 0; k < 2; ++k)
#pragma unroll
      for (int j = 0; j < 16; ++j) {
        const int K = 32 * k + 16 * hi + j;
        mA[k][j] = myM[n * C_IN + K];
      }

    // ---- phase 3: (msgs @ Wout) tile (8 WMMAs) + scatter-add ----
#pragma unroll
    for (int t = 0; t < 4; ++t) {
      v8f acc = {0.f, 0.f, 0.f, 0.f, 0.f, 0.f, 0.f, 0.f};
      acc = __builtin_amdgcn_wmma_f32_16x16x32_bf16(
          false, mA[0], false, woB[0][t], (short)0, acc, false, false);
      acc = __builtin_amdgcn_wmma_f32_16x16x32_bf16(
          false, mA[1], false, woB[1][t], (short)0, acc, false, false);
#pragma unroll
      for (int v = 0; v < 8; ++v) {
        const int d = myDst[8 * hi + v];
        if (d >= 0) {
          // no-return f32 add -> GLOBAL_ATOMIC_ADD_F32 (STOREcnt), hits L2
          (void)__hip_atomic_fetch_add(&out[(size_t)d * C_OUT + 16 * t + n],
                                       acc[v], __ATOMIC_RELAXED,
                                       __HIP_MEMORY_SCOPE_AGENT);
        }
      }
    }
  }
}

extern "C" void kernel_launch(void* const* d_in, const int* in_sizes, int n_in,
                              void* d_out, int out_size, void* d_ws, size_t ws_size,
                              hipStream_t stream) {
  const float* x       = (const float*)d_in[0];
  const float* rel_pos = (const float*)d_in[1];
  const int*   src_idx = (const int*)d_in[2];
  const int*   dst_idx = (const int*)d_in[3];
  const float* W1      = (const float*)d_in[4];
  const float* b1      = (const float*)d_in[5];
  const float* W2      = (const float*)d_in[6];
  const float* b2      = (const float*)d_in[7];
  const float* Wout    = (const float*)d_in[8];
  const float* bout    = (const float*)d_in[9];
  float*       out     = (float*)d_out;

  const int n_nodes = in_sizes[0] / C_IN;
  const int n_edges = in_sizes[2];

  // 1) out = bout (broadcast), so edge kernel can scatter into it directly
  const int total = n_nodes * C_OUT;
  pcconv_init_out<<<(total + BLOCK - 1) / BLOCK, BLOCK, 0, stream>>>(out, bout, total);

  // 2) fused edge pipeline
  const int ntiles = (n_edges + 15) >> 4;
  int blocks = (ntiles + WPB - 1) / WPB;
  if (blocks > 1024) blocks = 1024;   // 8192 waves, ~6 tiles each
  pcconv_edge_fused<<<blocks, BLOCK, 0, stream>>>(
      x, rel_pos, src_idx, dst_idx, W1, b1, W2, b2, Wout, out, n_edges);
}
